// HopfieldAttention_12292196402136
// MI455X (gfx1250) — compile-verified, matrix operand order checked by reading, not run on hardware
//
#include <hip/hip_runtime.h>
#include <hip/hip_bf16.h>

typedef __attribute__((ext_vector_type(16))) __bf16 v16bf;
typedef __attribute__((ext_vector_type(8)))  __bf16 v8bf;
typedef __attribute__((ext_vector_type(4)))  __bf16 v4bf;
typedef __attribute__((ext_vector_type(8)))  float  v8f;
typedef __attribute__((ext_vector_type(4)))  float  v4f;

// ---------------------------------------------------------------------------
// Load the 16 bf16 elements a lane owns of a 16x32 (16-bit) WMMA operand from
// a row-major source.  Per the CDNA5 ISA layout a lane holds two contiguous
// runs of 8 along K: [base .. base+7] and [base+16 .. base+23], where
// base = (lane>=16 ? 8 : 0) is baked into the pointer by the caller.
// ---------------------------------------------------------------------------
__device__ __forceinline__ v16bf load16bf(const __bf16* __restrict__ p) {
  const v8bf* q = (const v8bf*)p;               // 16B aligned
  v8bf r0 = q[0];                               // k = 0..7   (relative)
  v8bf r1 = q[2];                               // k = 16..23 (relative)
  v16bf r;
#pragma unroll
  for (int j = 0; j < 8; ++j) { r[j] = r0[j]; r[8 + j] = r1[j]; }
  return r;
}

// D = A*B + C in split-bf16 (3 WMMAs: lo*hi + hi*lo + hi*hi)
__device__ __forceinline__ v8f wmma3(v16bf ah, v16bf al, v16bf bh, v16bf bl, v8f c) {
  c = __builtin_amdgcn_wmma_f32_16x16x32_bf16(false, al, false, bh, (short)0, c, false, false);
  c = __builtin_amdgcn_wmma_f32_16x16x32_bf16(false, ah, false, bl, (short)0, c, false, false);
  c = __builtin_amdgcn_wmma_f32_16x16x32_bf16(false, ah, false, bh, (short)0, c, false, false);
  return c;
}

// ---------------------------------------------------------------------------
// Phase 0: one-time elementwise split of f32 arrays into bf16 hi/lo pairs.
// Pure bandwidth (~42 MB total) -- removes all conversion VALU from the
// WMMA inner loops, where it was previously re-done 64-256x per element.
// ---------------------------------------------------------------------------
__global__ __launch_bounds__(256) void split_convert(
    const float* __restrict__ src, __bf16* __restrict__ hi,
    __bf16* __restrict__ lo, int n4) {
  int i = blockIdx.x * blockDim.x + threadIdx.x;
  if (i >= n4) return;
  v4f x = ((const v4f*)src)[i];
  v4bf h, l;
#pragma unroll
  for (int j = 0; j < 4; ++j) {
    __bf16 hh = (__bf16)x[j];                   // RNE to bf16
    h[j] = hh;
    l[j] = (__bf16)(x[j] - (float)hh);          // residual
  }
  ((v4bf*)hi)[i] = h;
  ((v4bf*)lo)[i] = l;
}

// ---------------------------------------------------------------------------
// Phase 1: Q = query @ W_Q^T, K = memory @ W_K^T   (both [8,512,1024])
// One wave per 16(L) x 64(E_out) output tile: A-operand loads amortized over
// 4 accumulators; inner loop is b128 loads + 12 WMMAs only.
// ---------------------------------------------------------------------------
__global__ __launch_bounds__(128) void hopfield_proj(
    const __bf16* __restrict__ Xqh, const __bf16* __restrict__ Xql,
    const __bf16* __restrict__ Xmh, const __bf16* __restrict__ Xml,
    const __bf16* __restrict__ Wqh, const __bf16* __restrict__ Wql,
    const __bf16* __restrict__ Wkh, const __bf16* __restrict__ Wkl,
    __bf16* __restrict__ Qhi, __bf16* __restrict__ Qlo,
    __bf16* __restrict__ Khi, __bf16* __restrict__ Klo) {
  const int lane   = threadIdx.x & 31;
  const int waveId = (blockIdx.x * blockDim.x + threadIdx.x) >> 5;
  const int eT     = waveId & 15;          // 16 tiles of 64 along E_out
  const int qT     = (waveId >> 4) & 31;   // 32 tiles of 16 along L
  const int b      = (waveId >> 9) & 7;    // 8 batches
  const int which  = waveId >> 12;         // 0 = Q path, 1 = K path

  const __bf16* Xh = which ? Xmh : Xqh;
  const __bf16* Xl = which ? Xml : Xql;
  const __bf16* Wh = which ? Wkh : Wqh;
  const __bf16* Wl = which ? Wkl : Wql;
  __bf16* Ohi = which ? Khi : Qhi;
  __bf16* Olo = which ? Klo : Qlo;

  const int r15 = lane & 15;
  const int hi8 = (lane >> 4) * 8;         // lane-group K base offset

  const size_t aOff = (size_t)(b * 512 + qT * 16 + r15) * 1024 + hi8;
  size_t bOff[4];
#pragma unroll
  for (int s = 0; s < 4; ++s)
    bOff[s] = (size_t)(eT * 64 + s * 16 + r15) * 1024 + hi8;

  v8f acc[4] = {};
  for (int c = 0; c < 32; ++c) {           // 1024 / 32 reduction chunks
    v16bf aH = load16bf(Xh + aOff + c * 32);
    v16bf aL = load16bf(Xl + aOff + c * 32);
#pragma unroll
    for (int s = 0; s < 4; ++s) {
      v16bf bH = load16bf(Wh + bOff[s] + c * 32);
      v16bf bL = load16bf(Wl + bOff[s] + c * 32);
      acc[s] = wmma3(aH, aL, bH, bL, acc[s]);
    }
  }

  // f32 C/D layout: lane<16 -> rows 0..7, lane>=16 -> rows 8..15; col = lane&15
  const int mBase = qT * 16 + hi8;
#pragma unroll
  for (int s = 0; s < 4; ++s) {
    const int col = eT * 64 + s * 16 + r15;
#pragma unroll
    for (int r = 0; r < 8; ++r) {
      size_t idx = (size_t)(b * 512 + mBase + r) * 1024 + col;
      float v  = acc[s][r];
      __bf16 h = (__bf16)v;
      Ohi[idx] = h;
      Olo[idx] = (__bf16)(v - (float)h);
    }
  }
}

// ---------------------------------------------------------------------------
// Phase 2: logits[b,a,h,q,k] = sum_d Q[a,h,q,d]*K[b,h,k,d] * 0.125
//          energies = logsumexp over q / 0.125 ; mean over a
// One wave per (b, h, 16-wide k tile); B operand (K tile) register-resident.
// ---------------------------------------------------------------------------
__global__ __launch_bounds__(128) void hopfield_attn(
    const __bf16* __restrict__ Qhi, const __bf16* __restrict__ Qlo,
    const __bf16* __restrict__ Khi, const __bf16* __restrict__ Klo,
    float* __restrict__ out) {
  const int lane   = threadIdx.x & 31;
  const int waveId = (blockIdx.x * blockDim.x + threadIdx.x) >> 5;
  const int kT     = waveId & 31;          // 32 tiles of 16 along k
  const int h      = (waveId >> 5) & 15;   // 16 heads
  const int b      = (waveId >> 9) & 7;    // 8 memory batches

  const int r15 = lane & 15;
  const int hi8 = (lane >> 4) * 8;

  // B operand: K rows (columns of B), contiguous over d.  Held in registers.
  const size_t kBase = (size_t)(b * 512 + kT * 16 + r15) * 1024 + h * 64 + hi8;
  v16bf bH[2], bL[2];
#pragma unroll
  for (int c = 0; c < 2; ++c) {
    bH[c] = load16bf(Khi + kBase + c * 32);
    bL[c] = load16bf(Klo + kBase + c * 32);
  }

  const float SCALE = 0.125f;              // 64^-0.5
  float meanAcc = 0.0f;

  for (int a = 0; a < 8; ++a) {
    float m = -__builtin_inff();
    float s = 0.0f;
    const size_t qRowBase = (size_t)(a * 512 + r15) * 1024 + h * 64 + hi8;

    for (int qt = 0; qt < 32; ++qt) {
      const size_t qBase = qRowBase + (size_t)qt * (16 * 1024);
      __builtin_prefetch(Qhi + qBase + 16 * 1024, 0, 0);   // next q tile

      v8f acc = {};
#pragma unroll
      for (int c = 0; c < 2; ++c) {
        v16bf aH = load16bf(Qhi + qBase + c * 32);
        v16bf aL = load16bf(Qlo + qBase + c * 32);
        acc = wmma3(aH, aL, bH[c], bL[c], acc);
      }

      // Online logsumexp over the 8 q-rows this lane holds for its k column.
      float v[8], tmax = -__builtin_inff();
#pragma unroll
      for (int r = 0; r < 8; ++r) { v[r] = acc[r] * SCALE; tmax = fmaxf(tmax, v[r]); }
      float nm = fmaxf(m, tmax);
      float ts = 0.0f;
#pragma unroll
      for (int r = 0; r < 8; ++r) ts += __expf(v[r] - nm);
      s = s * __expf(m - nm) + ts;
      m = nm;
    }

    // Column k lives in lanes n and n+16 -> combine the two partials.
    float om = __shfl_xor(m, 16, 32);
    float os = __shfl_xor(s, 16, 32);
    float M  = fmaxf(m, om);
    float S  = s * __expf(m - M) + os * __expf(om - M);
    meanAcc += (M + __logf(S)) * 8.0f;     // / SCALE
  }

  if (lane < 16) {
    const int k = kT * 16 + r15;
    out[(size_t)b * (16 * 512) + h * 512 + k] = meanAcc * 0.125f;  // mean over a
  }
}

// ---------------------------------------------------------------------------
extern "C" void kernel_launch(void* const* d_in, const int* in_sizes, int n_in,
                              void* d_out, int out_size, void* d_ws, size_t ws_size,
                              hipStream_t stream) {
  const float* query = (const float*)d_in[0];   // [8,512,1024]
  const float* mem   = (const float*)d_in[1];   // [8,512,1024]
  const float* W_Q   = (const float*)d_in[2];   // [1024,1024]
  const float* W_K   = (const float*)d_in[3];   // [1024,1024]
  float* out = (float*)d_out;                   // [8,16,512]

  const size_t NA = (size_t)8 * 512 * 1024;     // 4 Mi activations
  const size_t NW = (size_t)1024 * 1024;        // 1 Mi weights

  __bf16* p   = (__bf16*)d_ws;                  // 72 MB total, L2 resident
  __bf16* Xqh = p;            __bf16* Xql = Xqh + NA;
  __bf16* Xmh = Xql + NA;     __bf16* Xml = Xmh + NA;
  __bf16* Wqh = Xml + NA;     __bf16* Wql = Wqh + NW;
  __bf16* Wkh = Wql + NW;     __bf16* Wkl = Wkh + NW;
  __bf16* Qhi = Wkl + NW;     __bf16* Qlo = Qhi + NA;
  __bf16* Khi = Qlo + NA;     __bf16* Klo = Khi + NA;

  // Phase 0: one-time f32 -> bf16 hi/lo split of all operands
  split_convert<<<(int)(NA / 4 / 256), 256, 0, stream>>>(query, Xqh, Xql, (int)(NA / 4));
  split_convert<<<(int)(NA / 4 / 256), 256, 0, stream>>>(mem,   Xmh, Xml, (int)(NA / 4));
  split_convert<<<(int)(NW / 4 / 256), 256, 0, stream>>>(W_Q,   Wqh, Wql, (int)(NW / 4));
  split_convert<<<(int)(NW / 4 / 256), 256, 0, stream>>>(W_K,   Wkh, Wkl, (int)(NW / 4));

  // Phase 1: 2 * 8 * 32 * 16 = 8192 waves, 4 waves / block
  hopfield_proj<<<2048, 128, 0, stream>>>(Xqh, Xql, Xmh, Xml, Wqh, Wql, Wkh, Wkl,
                                          Qhi, Qlo, Khi, Klo);
  // Phase 2: 8 * 16 * 32 = 4096 waves, 4 waves / block
  hopfield_attn<<<1024, 128, 0, stream>>>(Qhi, Qlo, Khi, Klo, out);
}
